// Painter_Varient_83992380440891
// MI455X (gfx1250) — matrix operand order repaired and checked by reading, not compiled, not run on hardware
//
#include <hip/hip_runtime.h>
#include <hip/hip_fp16.h>
#include <math.h>

// ---------------------------------------------------------------------------
// ViT block (SAM/Painter) forward for MI455X (gfx1250, wave32, WMMA).
// ~220 GFLOP, ~200MB unique HBM traffic -> compute bound: everything runs on
// v_wmma_f32_16x16x32_f16. Attention is flash-style (online softmax) so the
// 314MB [128,784,784] score tensor is never materialized; decomposed rel-pos
// bias becomes two 16x28 per-query-tile LDS tables.
// ---------------------------------------------------------------------------

typedef __attribute__((ext_vector_type(16))) _Float16 v16h;
typedef __attribute__((ext_vector_type(8)))  float    v8f;

#define N_ROWS   6272      // B*H*W = 8*28*28
#define CDIM     1024
#define TCDIM    3072
#define S_LEN    784
#define NHEAD    16
#define HDIM     64
#define HID_DIM  4096
#define BC_DIM   512
#define K_CONV2  4608      // 9*512
#define ATT_SCALE 0.125f   // 64^-0.5

__device__ inline v8f wmma16(v16h a, v16h b, v8f c) {
  // v_wmma_f32_16x16x32_f16: (neg_a, A, neg_b, B, c_mod, C, reuse_a, reuse_b)
  return __builtin_amdgcn_wmma_f32_16x16x32_f16(false, a, false, b, (short)0, c,
                                                false, false);
}

__device__ inline float gelu_f(float x) {
  return 0.5f * x * (1.0f + erff(x * 0.70710678118654752f));
}

// --- fragment loaders (wave32 layouts per CDNA5 ISA 7.12.2) -----
// A 16x32 f16: lane m=lane&15; lanes 0-15 hold K 0..7,16..23; lanes 16-31
// hold K 8..15,24..31 (elements e<8 -> K=kb+e; e>=8 -> K=kb+8+e).
__device__ inline v16h load_A16x32(const _Float16* rowbase, int ld) {
  const int lane = threadIdx.x & 31;
  const int m    = lane & 15;
  const int kb   = (lane & 16) ? 8 : 0;
  const _Float16* p = rowbase + (size_t)m * ld + kb;
  v16h r;
  *((uint4*)&r)       = *(const uint4*)(p);        // e0..7  -> K kb..kb+7
  *(((uint4*)&r) + 1) = *(const uint4*)(p + 16);   // e8..15 -> K kb+16..kb+23
  return r;
}

// B 32x16 f16 (row-major KxN tile): lane n=lane&15, elements = K in A's order.
__device__ inline v16h load_B32x16(const _Float16* base, int ld) {
  const int lane = threadIdx.x & 31;
  const int n    = lane & 15;
  const int kb   = (lane & 16) ? 8 : 0;
  v16h r;
#pragma unroll
  for (int s = 0; s < 16; ++s) r[s] = base[(size_t)(kb + (s < 8 ? s : s + 8)) * ld + n];
  return r;
}

// B fragment from K stored row-major [key][d]: lane n=key0+(lane&15), elements = d in A's order
__device__ inline v16h load_BT64(const _Float16* base) {
  return load_A16x32(base, 64);
}

// V fragment: lane n=dc+(lane&15), elements = keys kt+k in A's order (clamped on remainder tile)
__device__ inline v16h load_BV(const _Float16* vb, int kt, int dc) {
  const int lane = threadIdx.x & 31;
  const int n    = lane & 15;
  const int kb   = (lane & 16) ? 8 : 0;
  v16h r;
#pragma unroll
  for (int s = 0; s < 16; ++s) {
    int key = kt + kb + (s < 8 ? s : s + 8);
    if (key > S_LEN - 1) key = S_LEN - 1;
    r[s] = vb[(size_t)key * HDIM + dc + n];
  }
  return r;
}

// ---------------------------------------------------------------------------
// fp32 -> fp16 cast (weights once per launch; also x2 -> conv input)
// ---------------------------------------------------------------------------
__global__ void cast_f2h_kernel(const float* __restrict__ s,
                                _Float16* __restrict__ d, size_t n) {
  size_t i = (size_t)blockIdx.x * blockDim.x + threadIdx.x;
  if (i < n) d[i] = (_Float16)s[i];
}

// ---------------------------------------------------------------------------
// LayerNorm over last dim -> fp16 (optional fused GELU)
// ---------------------------------------------------------------------------
__global__ __launch_bounds__(256) void ln_f16_kernel(
    const float* __restrict__ x, const float* __restrict__ w,
    const float* __restrict__ b, _Float16* __restrict__ out,
    int C, float eps, int do_gelu) {
  __shared__ float rs[256], rq[256];
  const int row = blockIdx.x;
  const float* xr = x + (size_t)row * C;
  float s = 0.f, q = 0.f;
  for (int c = threadIdx.x; c < C; c += 256) { float v = xr[c]; s += v; q += v * v; }
  rs[threadIdx.x] = s; rq[threadIdx.x] = q;
  __syncthreads();
  for (int off = 128; off > 0; off >>= 1) {
    if ((int)threadIdx.x < off) {
      rs[threadIdx.x] += rs[threadIdx.x + off];
      rq[threadIdx.x] += rq[threadIdx.x + off];
    }
    __syncthreads();
  }
  const float mean = rs[0] / C;
  const float inv  = rsqrtf(rq[0] / C - mean * mean + eps);
  _Float16* outr = out + (size_t)row * C;
  for (int c = threadIdx.x; c < C; c += 256) {
    float v = (xr[c] - mean) * inv * w[c] + b[c];
    if (do_gelu) v = gelu_f(v);
    outr[c] = (_Float16)v;
  }
}

// final: out = residual + LN(conv3out)   (fp32 out == d_out)
__global__ __launch_bounds__(256) void ln_add_out_kernel(
    const float* __restrict__ x, const float* __restrict__ res,
    const float* __restrict__ w, const float* __restrict__ b,
    float* __restrict__ out, int C, float eps) {
  __shared__ float rs[256], rq[256];
  const int row = blockIdx.x;
  const float* xr = x + (size_t)row * C;
  float s = 0.f, q = 0.f;
  for (int c = threadIdx.x; c < C; c += 256) { float v = xr[c]; s += v; q += v * v; }
  rs[threadIdx.x] = s; rq[threadIdx.x] = q;
  __syncthreads();
  for (int off = 128; off > 0; off >>= 1) {
    if ((int)threadIdx.x < off) {
      rs[threadIdx.x] += rs[threadIdx.x + off];
      rq[threadIdx.x] += rq[threadIdx.x + off];
    }
    __syncthreads();
  }
  const float mean = rs[0] / C;
  const float inv  = rsqrtf(rq[0] / C - mean * mean + eps);
  for (int c = threadIdx.x; c < C; c += 256) {
    size_t idx = (size_t)row * C + c;
    out[idx] = res[idx] + ((xr[c] - mean) * inv * w[c] + b[c]);
  }
}

// ---------------------------------------------------------------------------
// WMMA GEMM: C[M,N] = A[M,K](f16) @ B[K,N](f16) (+bias) (+residual/gelu)
// 256 threads (8 waves), tile 128x64x32; wave -> 32x32 (2x2 fragments).
// epilogue: 0 = f32 out (+bias), 1 = f32 out + bias + residual,
//           2 = f16 out gelu(acc+bias), 3 = f16 out (acc+bias)
// ---------------------------------------------------------------------------
__global__ __launch_bounds__(256) void gemm_wmma_kernel(
    const _Float16* __restrict__ A, const _Float16* __restrict__ Bw,
    const float* __restrict__ bias, int M, int K, int Nn, int epilogue,
    const float* __restrict__ res, float* __restrict__ outF,
    _Float16* __restrict__ outH) {
  __shared__ _Float16 sA[128 * 40];   // 32-col tile padded to 40 halves
  __shared__ _Float16 sB[32 * 72];    // 64-col tile padded to 72 halves
  const int tid  = threadIdx.x;
  const int wave = tid >> 5;
  const int wm   = wave & 3;          // row group (4 x 32 rows)
  const int wn   = wave >> 2;         // col group (2 x 32 cols)
  const int bm   = blockIdx.x * 128;
  const int bn   = blockIdx.y * 64;

  const int ar = tid >> 1, ac = (tid & 1) * 16;   // A: 16 halves/thread
  const int br = tid >> 3, bc = (tid & 7) * 8;    // B: 8 halves/thread

  v8f acc[2][2] = {};

  for (int k0 = 0; k0 < K; k0 += 32) {
    const uint4* ga = (const uint4*)(A + (size_t)(bm + ar) * K + k0 + ac);
    uint4 a0 = ga[0], a1 = ga[1];
    const uint4* gb = (const uint4*)(Bw + (size_t)(k0 + br) * Nn + bn + bc);
    uint4 b0 = gb[0];
    if (k0 + 32 < K) {
      __builtin_prefetch(A + (size_t)(bm + ar) * K + k0 + 32 + ac, 0, 0);
      __builtin_prefetch(Bw + (size_t)(k0 + 32 + br) * Nn + bn + bc, 0, 0);
    }
    __syncthreads();
    *(uint4*)(sA + ar * 40 + ac)     = a0;
    *(uint4*)(sA + ar * 40 + ac + 8) = a1;
    *(uint4*)(sB + br * 72 + bc)     = b0;
    __syncthreads();

    v16h af0 = load_A16x32(sA + (size_t)(wm * 32 + 0)  * 40, 40);
    v16h af1 = load_A16x32(sA + (size_t)(wm * 32 + 16) * 40, 40);
    v16h bf0 = load_B32x16(sB + wn * 32 + 0, 72);
    v16h bf1 = load_B32x16(sB + wn * 32 + 16, 72);
    acc[0][0] = wmma16(af0, bf0, acc[0][0]);
    acc[0][1] = wmma16(af0, bf1, acc[0][1]);
    acc[1][0] = wmma16(af1, bf0, acc[1][0]);
    acc[1][1] = wmma16(af1, bf1, acc[1][1]);
  }

  const int lane = tid & 31;
  const int n    = lane & 15;
  const int rofs = (lane & 16) ? 8 : 0;
#pragma unroll
  for (int i = 0; i < 2; ++i)
#pragma unroll
    for (int j = 0; j < 2; ++j) {
      const int col = bn + wn * 32 + j * 16 + n;
      const float bv = bias ? bias[col] : 0.f;
#pragma unroll
      for (int r = 0; r < 8; ++r) {
        const int row = bm + wm * 32 + i * 16 + rofs + r;
        size_t idx = (size_t)row * Nn + col;
        float v = acc[i][j][r] + bv;
        if (epilogue == 1) v += res[idx];
        if (epilogue == 2) v = gelu_f(v);
        if (epilogue >= 2) outH[idx] = (_Float16)v;
        else               outF[idx] = v;
      }
    }
}

// ---------------------------------------------------------------------------
// 3x3 conv (pad 1) as implicit GEMM: M=6272 pixels, K=4608, N=512.
// K chunks of 32 never straddle a (ky,kx) boundary (512 % 32 == 0).
// ---------------------------------------------------------------------------
__global__ __launch_bounds__(256) void conv3x3_wmma_kernel(
    const _Float16* __restrict__ X,   // [B,28,28,512] f16
    const _Float16* __restrict__ Wt,  // [4608,512]    f16 (HWIO flattened)
    float* __restrict__ outF) {       // [B,28,28,512] f32
  __shared__ _Float16 sA[128 * 40];
  __shared__ _Float16 sB[32 * 72];
  const int tid  = threadIdx.x;
  const int wave = tid >> 5;
  const int wm   = wave & 3;
  const int wn   = wave >> 2;
  const int bm   = blockIdx.x * 128;
  const int bn   = blockIdx.y * 64;

  const int ar = tid >> 1, ac = (tid & 1) * 16;
  const int br = tid >> 3, bc = (tid & 7) * 8;

  const int P   = bm + ar;
  const int b   = P / 784;
  const int rem = P % 784;
  const int oh  = rem / 28, ow = rem % 28;

  v8f acc[2][2] = {};

  for (int k0 = 0; k0 < K_CONV2; k0 += 32) {
    const int g    = k0 >> 9;        // (ky*3+kx)
    const int cin0 = k0 & 511;
    const int ky = g / 3, kx = g % 3;
    const int ih = oh + ky - 1, iw = ow + kx - 1;
    uint4 a0 = {0, 0, 0, 0}, a1 = {0, 0, 0, 0};
    if ((unsigned)ih < 28u && (unsigned)iw < 28u) {
      const uint4* ga = (const uint4*)(X + (((size_t)b * 28 + ih) * 28 + iw) * 512
                                         + cin0 + ac);
      a0 = ga[0]; a1 = ga[1];
    }
    const uint4* gb = (const uint4*)(Wt + (size_t)(k0 + br) * 512 + bn + bc);
    uint4 b0 = gb[0];
    __syncthreads();
    *(uint4*)(sA + ar * 40 + ac)     = a0;
    *(uint4*)(sA + ar * 40 + ac + 8) = a1;
    *(uint4*)(sB + br * 72 + bc)     = b0;
    __syncthreads();

    v16h af0 = load_A16x32(sA + (size_t)(wm * 32 + 0)  * 40, 40);
    v16h af1 = load_A16x32(sA + (size_t)(wm * 32 + 16) * 40, 40);
    v16h bf0 = load_B32x16(sB + wn * 32 + 0, 72);
    v16h bf1 = load_B32x16(sB + wn * 32 + 16, 72);
    acc[0][0] = wmma16(af0, bf0, acc[0][0]);
    acc[0][1] = wmma16(af0, bf1, acc[0][1]);
    acc[1][0] = wmma16(af1, bf0, acc[1][0]);
    acc[1][1] = wmma16(af1, bf1, acc[1][1]);
  }

  const int lane = tid & 31;
  const int n    = lane & 15;
  const int rofs = (lane & 16) ? 8 : 0;
#pragma unroll
  for (int i = 0; i < 2; ++i)
#pragma unroll
    for (int j = 0; j < 2; ++j) {
      const int col = bn + wn * 32 + j * 16 + n;
#pragma unroll
      for (int r = 0; r < 8; ++r) {
        const int row = bm + wm * 32 + i * 16 + rofs + r;
        outF[(size_t)row * 512 + col] = acc[i][j][r];
      }
    }
}

// ---------------------------------------------------------------------------
// split fp32 qkv [N,3072] -> fp16 q/k/v, each [B*NH, S, HD]
// ---------------------------------------------------------------------------
__global__ void qkv_split_kernel(const float* __restrict__ qkv,
                                 _Float16* __restrict__ q,
                                 _Float16* __restrict__ k,
                                 _Float16* __restrict__ v) {
  size_t gid = (size_t)blockIdx.x * blockDim.x + threadIdx.x;
  const size_t total = (size_t)N_ROWS * TCDIM;
  if (gid >= total) return;
  const int row   = (int)(gid / TCDIM);
  const int c     = (int)(gid % TCDIM);
  const int which = c / CDIM;
  const int cc    = c % CDIM;
  const int nh = cc / HDIM, d = cc % HDIM;
  const int b = row / S_LEN, s = row % S_LEN;
  const size_t dst = (((size_t)(b * NHEAD + nh)) * S_LEN + s) * HDIM + d;
  const _Float16 val = (_Float16)qkv[gid];
  if (which == 0)      q[dst] = val;
  else if (which == 1) k[dst] = val;
  else                 v[dst] = val;
}

// ---------------------------------------------------------------------------
// Flash attention with decomposed rel-pos bias.
// grid (13, 128): blockIdx.y = head (b*16+nh); 4 waves/block, one 16-query
// tile per wave. Streams keys in 32-wide tiles (24 full + one 16 remainder),
// online softmax in fp32, P@V via WMMA with P staged through per-wave LDS.
// Bias for query (qh,qw) vs key (kh,kw) = bh[m][kh] + bw[m][kw], precomputed
// from *unscaled* q (reference scales only the q@k^T term).
// ---------------------------------------------------------------------------
__global__ __launch_bounds__(128) void flash_attn_kernel(
    const _Float16* __restrict__ Q, const _Float16* __restrict__ Kt,
    const _Float16* __restrict__ V, const float* __restrict__ rel_h,
    const float* __restrict__ rel_w, _Float16* __restrict__ Out) {
  __shared__ float    sBH[4][16][28];
  __shared__ float    sBW[4][16][28];
  __shared__ _Float16 sP[4][16 * 40];

  const int tid  = threadIdx.x;
  const int wave = tid >> 5;
  const int lane = tid & 31;
  const int head = blockIdx.y;                 // b*16 + nh
  const int qt   = blockIdx.x * 4 + wave;
  if (qt >= 49) return;                        // wave-uniform (EXEC stays all-1)
  const int b  = head >> 4;
  const int nh = head & 15;
  const int q0 = qt * 16;

  const _Float16* qbase = Q  + ((size_t)head * S_LEN + q0) * HDIM;
  const _Float16* kbase = Kt + (size_t)head * S_LEN * HDIM;
  const _Float16* vbase = V  + (size_t)head * S_LEN * HDIM;

  // ---- per-query-tile bias tables: bh[m][kh], bw[m][kw] ----
  for (int idx = lane; idx < 16 * 28; idx += 32) {
    const int m = idx / 28, t = idx % 28;
    const int sidx = q0 + m;
    const int qh = sidx / 28, qw = sidx % 28;
    const _Float16* qp = qbase + (size_t)m * HDIM;
    const float* rh = rel_h + (size_t)(qh - t + 27) * HDIM;
    const float* rw = rel_w + (size_t)(qw - t + 27) * HDIM;
    float ah = 0.f, aw = 0.f;
#pragma unroll 8
    for (int d = 0; d < HDIM; ++d) {
      const float qd = (float)qp[d];
      ah += qd * rh[d];
      aw += qd * rw[d];
    }
    sBH[wave][m][t] = ah;
    sBW[wave][m][t] = aw;
  }
  asm volatile("s_wait_dscnt 0" ::: "memory");

  const v16h aq0 = load_A16x32(qbase, HDIM);        // d 0..31
  const v16h aq1 = load_A16x32(qbase + 32, HDIM);   // d 32..63

  v8f o0 = {}, o1 = {}, o2 = {}, o3 = {};
  float rm[8], rl[8];
#pragma unroll
  for (int r = 0; r < 8; ++r) { rm[r] = -1e30f; rl[r] = 0.f; }

  const int n  = lane & 15;
  const int hi = (lane >> 4) & 1;
  _Float16* pS = &sP[wave][0];

  for (int kt = 0; kt < S_LEN; kt += 32) {
    const bool full = (kt + 32 <= S_LEN);
    v8f s0 = {}, s1 = {};
    {
      v16h bk = load_BT64(kbase + (size_t)kt * HDIM + 0);
      s0 = wmma16(aq0, bk, s0);
      bk = load_BT64(kbase + (size_t)kt * HDIM + 32);
      s0 = wmma16(aq1, bk, s0);
    }
    if (full) {
      v16h bk = load_BT64(kbase + (size_t)(kt + 16) * HDIM + 0);
      s1 = wmma16(aq0, bk, s1);
      bk = load_BT64(kbase + (size_t)(kt + 16) * HDIM + 32);
      s1 = wmma16(aq1, bk, s1);
    }

    // scale + bias (fragment col n is fixed per lane)
    const int key0g = kt + n;
    const int key1g = (kt + 16 + n < S_LEN) ? (kt + 16 + n) : (S_LEN - 1);
    const int kh0 = key0g / 28, kw0 = key0g % 28;
    const int kh1 = key1g / 28, kw1 = key1g % 28;
#pragma unroll
    for (int r = 0; r < 8; ++r) {
      const int m = r + hi * 8;
      s0[r] = s0[r] * ATT_SCALE + sBH[wave][m][kh0] + sBW[wave][m][kw0];
      if (full)
        s1[r] = s1[r] * ATT_SCALE + sBH[wave][m][kh1] + sBW[wave][m][kw1];
    }

    // online softmax: row stats live per (vgpr r, lane-half) -> width-16 xor
#pragma unroll
    for (int r = 0; r < 8; ++r) {
      float t = full ? fmaxf(s0[r], s1[r]) : s0[r];
      t = fmaxf(t, __shfl_xor(t, 1, 16));
      t = fmaxf(t, __shfl_xor(t, 2, 16));
      t = fmaxf(t, __shfl_xor(t, 4, 16));
      t = fmaxf(t, __shfl_xor(t, 8, 16));
      const float mn = fmaxf(rm[r], t);
      const float al = __expf(rm[r] - mn);
      rm[r] = mn;
      const float p0 = __expf(s0[r] - mn);
      const float p1 = full ? __expf(s1[r] - mn) : 0.f;
      s0[r] = p0;
      s1[r] = p1;
      float su = p0 + p1;
      su += __shfl_xor(su, 1, 16);
      su += __shfl_xor(su, 2, 16);
      su += __shfl_xor(su, 4, 16);
      su += __shfl_xor(su, 8, 16);
      rl[r] = rl[r] * al + su;
      o0[r] *= al; o1[r] *= al; o2[r] *= al; o3[r] *= al;
    }

    // stage P (C-layout -> LDS -> A-layout)
#pragma unroll
    for (int r = 0; r < 8; ++r) {
      const int m = r + hi * 8;
      pS[m * 40 + n]      = (_Float16)s0[r];
      pS[m * 40 + 16 + n] = (_Float16)s1[r];   // zeros on remainder tile
    }
    asm volatile("s_wait_dscnt 0" ::: "memory");
    const v16h ap = load_A16x32(pS, 40);

    o0 = wmma16(ap, load_BV(vbase, kt, 0),  o0);
    o1 = wmma16(ap, load_BV(vbase, kt, 16), o1);
    o2 = wmma16(ap, load_BV(vbase, kt, 32), o2);
    o3 = wmma16(ap, load_BV(vbase, kt, 48), o3);
  }

  // normalize + scatter to [b*784+s][nh*64+d] fp16 (feeds proj GEMM)
  _Float16* obase = Out + ((size_t)b * S_LEN + q0) * CDIM + nh * HDIM;
#pragma unroll
  for (int r = 0; r < 8; ++r) {
    const int m = r + hi * 8;
    const float inv = 1.0f / rl[r];
    _Float16* orow = obase + (size_t)m * CDIM;
    orow[0  + n] = (_Float16)(o0[r] * inv);
    orow[16 + n] = (_Float16)(o1[r] * inv);
    orow[32 + n] = (_Float16)(o2[r] * inv);
    orow[48 + n] = (_Float16)(o3[r] * inv);
  }
}

// ---------------------------------------------------------------------------
// host-side orchestration
// ---------------------------------------------------------------------------
extern "C" void kernel_launch(void* const* d_in, const int* in_sizes, int n_in,
                              void* d_out, int out_size, void* d_ws,
                              size_t ws_size, hipStream_t stream) {
  const float* x     = (const float*)d_in[0];
  const float* n1w   = (const float*)d_in[1];
  const float* n1b   = (const float*)d_in[2];
  const float* qkvW  = (const float*)d_in[3];
  const float* qkvB  = (const float*)d_in[4];
  const float* projW = (const float*)d_in[5];
  const float* projB = (const float*)d_in[6];
  const float* relH  = (const float*)d_in[7];
  const float* relW  = (const float*)d_in[8];
  const float* n2w   = (const float*)d_in[9];
  const float* n2b   = (const float*)d_in[10];
  const float* fc1W  = (const float*)d_in[11];
  const float* fc1B  = (const float*)d_in[12];
  const float* fc2W  = (const float*)d_in[13];
  const float* fc2B  = (const float*)d_in[14];
  const float* c1W   = (const float*)d_in[15];
  const float* l1w   = (const float*)d_in[16];
  const float* l1b   = (const float*)d_in[17];
  const float* c2W   = (const float*)d_in[18];
  const float* l2w   = (const float*)d_in[19];
  const float* l2b   = (const float*)d_in[20];
  const float* c3W   = (const float*)d_in[21];
  const float* l3w   = (const float*)d_in[22];
  const float* l3b   = (const float*)d_in[23];
  (void)in_sizes; (void)n_in; (void)out_size; (void)ws_size;

  // ---- workspace bump allocator (~310 MB) ----
  char* wp = (char*)d_ws;
  auto alloc = [&](size_t bytes) {
    char* p = wp;
    wp += (bytes + 255) & ~(size_t)255;
    return p;
  };
  const size_t NC = (size_t)N_ROWS * CDIM;

  _Float16* h_qkvW = (_Float16*)alloc((size_t)CDIM * TCDIM * 2);
  _Float16* h_projW = (_Float16*)alloc((size_t)CDIM * CDIM * 2);
  _Float16* h_fc1W = (_Float16*)alloc((size_t)CDIM * HID_DIM * 2);
  _Float16* h_fc2W = (_Float16*)alloc((size_t)HID_DIM * CDIM * 2);
  _Float16* h_c1W  = (_Float16*)alloc((size_t)CDIM * BC_DIM * 2);
  _Float16* h_c2W  = (_Float16*)alloc((size_t)K_CONV2 * BC_DIM * 2);
  _Float16* h_c3W  = (_Float16*)alloc((size_t)BC_DIM * CDIM * 2);
  _Float16* h_xn   = (_Float16*)alloc(NC * 2);              // LN1 / LN2 out
  _Float16* h_q    = (_Float16*)alloc(NC * 2);
  _Float16* h_k    = (_Float16*)alloc(NC * 2);
  _Float16* h_v    = (_Float16*)alloc(NC * 2);
  _Float16* h_attn = (_Float16*)alloc(NC * 2);
  _Float16* h_hid  = (_Float16*)alloc((size_t)N_ROWS * HID_DIM * 2);
  _Float16* h_xres = (_Float16*)alloc(NC * 2);              // f16(x2) for conv1
  _Float16* h_bn1  = (_Float16*)alloc((size_t)N_ROWS * BC_DIM * 2);
  _Float16* h_bn2  = (_Float16*)alloc((size_t)N_ROWS * BC_DIM * 2);
  float* f_qkv = (float*)alloc((size_t)N_ROWS * TCDIM * 4);
  float* f_x1  = (float*)alloc(NC * 4);
  float* f_x2  = (float*)alloc(NC * 4);
  float* f_bn  = f_qkv;                       // conv1/conv2 f32 out (reuse)
  float* f_c3  = f_qkv + NC;                  // conv3 f32 out (disjoint reuse)

  auto cast = [&](const float* s, _Float16* d, size_t n) {
    cast_f2h_kernel<<<dim3((unsigned)((n + 255) / 256)), dim3(256), 0, stream>>>(s, d, n);
  };
  auto gemm = [&](const _Float16* A, const _Float16* Bw, const float* bias,
                  int M, int K, int Nn, int ep, const float* res, float* oF,
                  _Float16* oH) {
    gemm_wmma_kernel<<<dim3(M / 128, Nn / 64), dim3(256), 0, stream>>>(
        A, Bw, bias, M, K, Nn, ep, res, oF, oH);
  };

  // ---- weights fp32 -> fp16 (deterministic; recomputed each launch) ----
  cast(qkvW,  h_qkvW, (size_t)CDIM * TCDIM);
  cast(projW, h_projW, (size_t)CDIM * CDIM);
  cast(fc1W,  h_fc1W, (size_t)CDIM * HID_DIM);
  cast(fc2W,  h_fc2W, (size_t)HID_DIM * CDIM);
  cast(c1W,   h_c1W,  (size_t)CDIM * BC_DIM);
  cast(c2W,   h_c2W,  (size_t)K_CONV2 * BC_DIM);
  cast(c3W,   h_c3W,  (size_t)BC_DIM * CDIM);

  // ---- attention branch ----
  ln_f16_kernel<<<dim3(N_ROWS), dim3(256), 0, stream>>>(x, n1w, n1b, h_xn,
                                                        CDIM, 1e-5f, 0);
  gemm(h_xn, h_qkvW, qkvB, N_ROWS, CDIM, TCDIM, 0, nullptr, f_qkv, nullptr);
  {
    size_t tot = (size_t)N_ROWS * TCDIM;
    qkv_split_kernel<<<dim3((unsigned)((tot + 255) / 256)), dim3(256), 0,
                       stream>>>(f_qkv, h_q, h_k, h_v);
  }
  flash_attn_kernel<<<dim3(13, 128), dim3(128), 0, stream>>>(h_q, h_k, h_v,
                                                             relH, relW, h_attn);
  // x1 = x + attn_out @ proj_w + proj_b
  gemm(h_attn, h_projW, projB, N_ROWS, CDIM, CDIM, 1, x, f_x1, nullptr);

  // ---- MLP branch ----
  ln_f16_kernel<<<dim3(N_ROWS), dim3(256), 0, stream>>>(f_x1, n2w, n2b, h_xn,
                                                        CDIM, 1e-5f, 0);
  gemm(h_xn, h_fc1W, fc1B, N_ROWS, CDIM, HID_DIM, 2, nullptr, nullptr, h_hid);
  gemm(h_hid, h_fc2W, fc2B, N_ROWS, HID_DIM, CDIM, 1, f_x1, f_x2, nullptr);

  // ---- bottleneck conv branch ----
  cast(f_x2, h_xres, NC);
  gemm(h_xres, h_c1W, nullptr, N_ROWS, CDIM, BC_DIM, 0, nullptr, f_bn, nullptr);
  ln_f16_kernel<<<dim3(N_ROWS), dim3(256), 0, stream>>>(f_bn, l1w, l1b, h_bn1,
                                                        BC_DIM, 1e-6f, 1);
  conv3x3_wmma_kernel<<<dim3(49, 8), dim3(256), 0, stream>>>(h_bn1, h_c2W, f_bn);
  ln_f16_kernel<<<dim3(N_ROWS), dim3(256), 0, stream>>>(f_bn, l2w, l2b, h_bn2,
                                                        BC_DIM, 1e-6f, 1);
  gemm(h_bn2, h_c3W, nullptr, N_ROWS, BC_DIM, CDIM, 0, nullptr, f_c3, nullptr);
  ln_add_out_kernel<<<dim3(N_ROWS), dim3(256), 0, stream>>>(
      f_c3, f_x2, l3w, l3b, (float*)d_out, CDIM, 1e-6f);
}
